// AttentionHead_27290222199233
// MI455X (gfx1250) — compile-verified, matrix operand order checked by reading, not run on hardware
//
#include <hip/hip_runtime.h>

// ---------------- problem constants ----------------
constexpr int Bn = 8;      // batch
constexpr int Sn = 2048;   // sequence
constexpr int En = 1024;   // embed
constexpr int Dn = 128;    // DQ == DV
// scores / sqrt(DQ) folded with log2(e): softmax via exp2 (v_exp_f32)
constexpr float SCALE_L2E = 0.08838834764831845f * 1.44269504088896340f;

typedef __attribute__((ext_vector_type(16))) _Float16 v16h;
typedef __attribute__((ext_vector_type(8)))  _Float16 v8h;
typedef __attribute__((ext_vector_type(4)))  _Float16 v4h;
typedef __attribute__((ext_vector_type(2)))  _Float16 v2h;
typedef __attribute__((ext_vector_type(8)))  float    v8f;
typedef __attribute__((ext_vector_type(4)))  float    v4f;
typedef unsigned int uint32x4 __attribute__((ext_vector_type(4)));
typedef int          int32x4  __attribute__((ext_vector_type(4)));
typedef int          int32x8  __attribute__((ext_vector_type(8)));

// Tensor Data Mover path, guarded so a missing builtin degrades gracefully.
#if defined(__gfx1250__) && __has_builtin(__builtin_amdgcn_tensor_load_to_lds) && \
    __has_builtin(__builtin_amdgcn_s_wait_tensorcnt)
#define USE_TDM 1
#else
#define USE_TDM 0
#endif

// A/B fragment (16x32 f16): per-lane halves are two contiguous 8-half runs
// at +0 and +16 halves (CDNA5 ISA 7.12.2) -> two b128 loads.
__device__ __forceinline__ v16h load_frag_h(const _Float16* p) {
    v8h lo = *(const v8h*)p;
    v8h hi = *(const v8h*)(p + 16);
    return __builtin_shufflevector(lo, hi, 0,1,2,3,4,5,6,7,8,9,10,11,12,13,14,15);
}
// Same pattern but f32 source (two 32B runs) with convert.
__device__ __forceinline__ v16h load_frag_f32(const float* p) {
    v8f lo = *(const v8f*)p;
    v8f hi = *(const v8f*)(p + 16);
    v16h r;
    #pragma unroll
    for (int i = 0; i < 8; ++i) { r[i] = (_Float16)lo[i]; r[8 + i] = (_Float16)hi[i]; }
    return r;
}

// =====================================================================
// Kernel 0: convert Wq|Wk|Wv (f32 [128,1024]) to f16 once.
// =====================================================================
__global__ __launch_bounds__(256)
void w_to_f16(const float* __restrict__ Wq, const float* __restrict__ Wk,
              const float* __restrict__ Wv, _Float16* __restrict__ Wh)
{
    const int z = blockIdx.y;
    const float* W = (z == 0) ? Wq : (z == 1) ? Wk : Wv;
    _Float16* O = Wh + (size_t)z * Dn * En;
    const int i = (blockIdx.x * 256 + threadIdx.x) * 4;
    v4f x = *(const v4f*)(W + i);
    v4h h;
    h[0] = (_Float16)x[0]; h[1] = (_Float16)x[1];
    h[2] = (_Float16)x[2]; h[3] = (_Float16)x[3];
    *(v4h*)(O + i) = h;
}

// =====================================================================
// Kernel 1: QKV projection, LDS-free.  out = X[16384,1024] x W^T + b.
// grid = (BS/16, 1, 3), block = 128 (4 waves); wave w owns cols [32w,32w+32).
// =====================================================================
__global__ __launch_bounds__(128)
void qkv_proj(const float* __restrict__ q_in, const float* __restrict__ k_in,
              const float* __restrict__ v_in,
              const _Float16* __restrict__ Wh,
              const float* __restrict__ bq, const float* __restrict__ bk,
              const float* __restrict__ bv,
              _Float16* __restrict__ qo, _Float16* __restrict__ ko,
              _Float16* __restrict__ vo)
{
    const int z = blockIdx.z;
    const float*    X    = (z == 0) ? q_in : (z == 1) ? k_in : v_in;
    const _Float16* Wz   = Wh + (size_t)z * Dn * En;
    const float*    bias = (z == 0) ? bq : (z == 1) ? bk : bv;
    _Float16*       O    = (z == 0) ? qo : (z == 1) ? ko : vo;

    const int rowBase = blockIdx.x * 16;
    const int tid  = threadIdx.x;
    const int wave = tid >> 5;
    const int lane = tid & 31;
    const int lrow = lane & 15;
    const int kb   = (lane >> 4) * 8;

    const int n0 = wave * 32 + lrow;
    const int n1 = n0 + 16;
    const float*    arow = X  + (size_t)(rowBase + lrow) * En;
    const _Float16* w0   = Wz + (size_t)n0 * En;
    const _Float16* w1   = Wz + (size_t)n1 * En;

    v8f c0 = {}, c1 = {};
    #pragma unroll 4
    for (int e0 = 0; e0 < En; e0 += 32) {
        v16h a  = load_frag_f32(arow + e0 + kb);
        v16h b0 = load_frag_h(w0 + e0 + kb);
        v16h b1 = load_frag_h(w1 + e0 + kb);
        c0 = __builtin_amdgcn_wmma_f32_16x16x32_f16(false, a, false, b0,
                                                    (short)0, c0, false, false);
        c1 = __builtin_amdgcn_wmma_f32_16x16x32_f16(false, a, false, b1,
                                                    (short)0, c1, false, false);
    }

    const float bias0 = bias[n0], bias1 = bias[n1];
    const int Mo = (lane >> 4) * 8;                // C layout: M = r + 8*(lane>=16)
    #pragma unroll
    for (int r = 0; r < 8; ++r) {
        const size_t row = (size_t)(rowBase + Mo + r);
        O[row * Dn + n0] = (_Float16)(c0[r] + bias0);
        O[row * Dn + n1] = (_Float16)(c1[r] + bias1);
    }
}

// =====================================================================
// Kernel 2: causal flash attention, double-buffered (software pipelined).
// grid = (S/64, B), block = 128 (4 waves); wave w owns q rows [64bx+16w,+16).
// Per step: wait_tensorcnt -> barrier -> issue stage(next) -> compute(cur).
// K tile via TDM when available. Key columns permuted {k,k+16}->{2k,2k+1}
// so P and V^T LDS stores are b32. Row-sums of P via ones-fragment WMMA.
// =====================================================================
__global__ __launch_bounds__(128)
void flash_attn(const _Float16* __restrict__ qf,
                const _Float16* __restrict__ kf,
                const _Float16* __restrict__ vf,
                float* __restrict__ out)
{
    const int b    = blockIdx.y;
    const int q0   = blockIdx.x * 64;
    const int tid  = threadIdx.x;
    const int wave = tid >> 5;
    const int lane = tid & 31;
    const int lrow = lane & 15;
    const int kbp  = (lane >> 4) * 8;
    const int Mo   = (lane >> 4) * 8;
    const int qRow0 = q0 + wave * 16;

    __shared__ __align__(16) _Float16 Kl[2][32 * 128];   // [buf][key][d]
    __shared__ __align__(16) _Float16 Vt[2][128 * 32];   // [buf][d][logical key]
    __shared__ __align__(16) _Float16 Pl[4][16 * 32];    // per-wave P relayout

    const _Float16* qb  = qf + (size_t)b * Sn * Dn;
    const _Float16* kgb = kf + (size_t)b * Sn * Dn;
    const _Float16* vgb = vf + (size_t)b * Sn * Dn;

    // Q fragments (4 d-chunks), constant across key loop
    v16h qa[4];
    {
        const _Float16* qrow = qb + (size_t)(qRow0 + lrow) * Dn;
        #pragma unroll
        for (int c = 0; c < 4; ++c) qa[c] = load_frag_h(qrow + c * 32 + kbp);
    }

    // all-ones B fragment: wmma(P, ones) -> per-row sums in C layout
    v16h ones;
    #pragma unroll
    for (int i = 0; i < 16; ++i) ones[i] = (_Float16)1.0f;
    const v8f vzero = {};

    v8f o[8] = {};
    v8f l = {};
    float m[8];
    #pragma unroll
    for (int r = 0; r < 8; ++r) m[r] = -1e30f;

    // V^T staging: thread owns true keys {kp, kp+16} -> logical cols {2kp,2kp+1}
    const int kp   = tid & 15;
    const int dblk = (tid >> 4) * 16;

#if USE_TDM
    // Loop-invariant part of the K-tile tensor descriptor (D#), per ISA 08 §8.
    int32x8 g1;
    g1[0] = 0x00010000;          // data_size = 2B; no multicast/pad/iterate
    g1[1] = (int)(128u << 16);   // tensor_dim0 = 128 (d, contiguous)
    g1[2] = (int)(2048u << 16);  // tensor_dim1 = 2048 (keys)
    g1[3] = (int)(128u << 16);   // tile_dim0 = 128
    g1[4] = 32;                  // tile_dim1 = 32 keys, tile_dim2 = 0
    g1[5] = 128;                 // tensor_dim0_stride = 128 elements
    g1[6] = 0;
    g1[7] = 0;
    const int32x4 z4 = {0, 0, 0, 0};
#if __clang_major__ >= 23
    const int32x8 z8 = {0, 0, 0, 0, 0, 0, 0, 0};
#endif
#endif

    // stage key tile [k0,k0+32) into buffer `sb`
    auto stage = [&](int k0, int sb) {
#if USE_TDM
        if (wave == 0) {   // one TDM op per block per step; EXEC-independent
            const unsigned long long ga =
                (unsigned long long)(size_t)(kgb + (size_t)k0 * Dn);
            uint32x4 g0;
            g0[0] = 1u;                                  // count=1, user mode
            g0[1] = (unsigned)(size_t)(&Kl[sb][0]);      // lds_addr
            g0[2] = (unsigned)ga;                        // global_addr[31:0]
            g0[3] = ((unsigned)(ga >> 32) & 0x01FFFFFFu) // global_addr[56:32]
                    | 0x80000000u;                       // type=2 ("image")
#if __clang_major__ >= 23
            __builtin_amdgcn_tensor_load_to_lds(g0, g1, z4, z4, z8, 0);
#else
            __builtin_amdgcn_tensor_load_to_lds(g0, g1, z4, z4, 0);
#endif
        }
#else
        #pragma unroll
        for (int it = 0; it < 4; ++it) {
            const int idx8 = tid + it * 128;
            const int kr = idx8 >> 4, d8 = (idx8 & 15) * 8;
            *(v8h*)&Kl[sb][kr * 128 + d8] =
                *(const v8h*)&kgb[(size_t)(k0 + kr) * Dn + d8];
        }
#endif
        // V^T: 4x b128 global loads, 16x b32 LDS stores
        const _Float16* va = &vgb[(size_t)(k0 + kp) * Dn + dblk];   // key kp
        const _Float16* vc = va + (size_t)16 * Dn;                  // key kp+16
        v8h a0 = *(const v8h*)va, a1 = *(const v8h*)(va + 8);
        v8h b0 = *(const v8h*)vc, b1 = *(const v8h*)(vc + 8);
        #pragma unroll
        for (int j = 0; j < 8; ++j) {
            v2h p0; p0[0] = a0[j]; p0[1] = b0[j];
            *(v2h*)&Vt[sb][(dblk + j) * 32 + 2 * kp] = p0;
            v2h p1; p1[0] = a1[j]; p1[1] = b1[j];
            *(v2h*)&Vt[sb][(dblk + 8 + j) * 32 + 2 * kp] = p1;
        }
    };

    const int kMax = q0 + 64;
    stage(0, 0);                                   // prologue
    for (int k0 = 0, buf = 0; k0 < kMax; k0 += 32, buf ^= 1) {
#if USE_TDM
        if (wave == 0) __builtin_amdgcn_s_wait_tensorcnt(0);
#endif
        __syncthreads();                           // current buffer ready
        if (k0 + 32 < kMax) stage(k0 + 32, buf ^ 1);   // overlap with compute

        // ---- scores: two 16x16 tiles over true keys [k0,+16) and [k0+16,+32) ----
        const _Float16* Kb = &Kl[buf][0];
        v8f s0 = {}, s1 = {};
        #pragma unroll
        for (int c = 0; c < 4; ++c) {
            v16h kb0 = load_frag_h(Kb + lrow * 128        + c * 32 + kbp);
            v16h kb1 = load_frag_h(Kb + (lrow + 16) * 128 + c * 32 + kbp);
            s0 = __builtin_amdgcn_wmma_f32_16x16x32_f16(false, qa[c], false, kb0,
                                                        (short)0, s0, false, false);
            s1 = __builtin_amdgcn_wmma_f32_16x16x32_f16(false, qa[c], false, kb1,
                                                        (short)0, s1, false, false);
        }

        // ---- causal mask + online softmax (exp2 domain) ----
        const int n0 = k0 + lrow;                  // true key of s0 column
        const int n1 = n0 + 16;                    // true key of s1 column
        const v8f s0s = s0 * SCALE_L2E;            // packed f32 muls
        const v8f s1s = s1 * SCALE_L2E;
        v8f av;                                    // per-row alpha
        #pragma unroll
        for (int r = 0; r < 8; ++r) {
            const int qr = qRow0 + Mo + r;
            const float e0s = (n0 <= qr) ? s0s[r] : -1e30f;
            const float e1s = (n1 <= qr) ? s1s[r] : -1e30f;

            float rm = fmaxf(e0s, e1s);
            rm = fmaxf(rm, __shfl_xor(rm, 1));
            rm = fmaxf(rm, __shfl_xor(rm, 2));
            rm = fmaxf(rm, __shfl_xor(rm, 4));
            rm = fmaxf(rm, __shfl_xor(rm, 8));

            const float mnew = fmaxf(m[r], rm);
            av[r] = __builtin_exp2f(m[r] - mnew);
            m[r] = mnew;

            const float p0 = __builtin_exp2f(e0s - mnew);
            const float p1 = __builtin_exp2f(e1s - mnew);

            // P (C layout) -> LDS, keys {n0,n1} at logical cols {2*lrow, 2*lrow+1}
            v2h ph; ph[0] = (_Float16)p0; ph[1] = (_Float16)p1;
            *(v2h*)&Pl[wave][(Mo + r) * 32 + 2 * lrow] = ph;
        }

        #pragma unroll
        for (int t = 0; t < 8; ++t) o[t] *= av;    // packed f32 rescale

        asm volatile("s_wait_dscnt 0" ::: "memory");   // same-wave LDS RAW

        v16h pa = load_frag_h(&Pl[wave][lrow * 32 + kbp]);

        // row sums of P via the matrix pipe (replaces 32 shuffles)
        v8f lsum = __builtin_amdgcn_wmma_f32_16x16x32_f16(false, pa, false, ones,
                                                          (short)0, vzero, false, false);
        l = l * av + lsum;                         // packed f32 fma

        // ---- O += P x V over 8 d-tiles ----
        const _Float16* Vb = &Vt[buf][0];
        #pragma unroll
        for (int t = 0; t < 8; ++t) {
            v16h vb = load_frag_h(Vb + (t * 16 + lrow) * 32 + kbp);
            o[t] = __builtin_amdgcn_wmma_f32_16x16x32_f16(false, pa, false, vb,
                                                          (short)0, o[t], false, false);
        }
    }

    // ---- epilogue: normalize and store ----
    v8f inv;
    #pragma unroll
    for (int r = 0; r < 8; ++r) inv[r] = 1.0f / l[r];
    #pragma unroll
    for (int t = 0; t < 8; ++t) {
        const v8f ov = o[t] * inv;                 // packed f32
        #pragma unroll
        for (int r = 0; r < 8; ++r) {
            const int qr = qRow0 + Mo + r;
            const int d  = t * 16 + lrow;
            out[((size_t)b * Sn + qr) * Dn + d] = ov[r];
        }
    }
}

// =====================================================================
extern "C" void kernel_launch(void* const* d_in, const int* in_sizes, int n_in,
                              void* d_out, int out_size, void* d_ws, size_t ws_size,
                              hipStream_t stream)
{
    (void)in_sizes; (void)n_in; (void)out_size; (void)ws_size;
    const float* query  = (const float*)d_in[0];
    const float* key_in = (const float*)d_in[1];
    const float* value  = (const float*)d_in[2];
    const float* Wq     = (const float*)d_in[3];
    const float* bq     = (const float*)d_in[4];
    const float* Wk     = (const float*)d_in[5];
    const float* bk     = (const float*)d_in[6];
    const float* Wv     = (const float*)d_in[7];
    const float* bv     = (const float*)d_in[8];
    float* out = (float*)d_out;

    // workspace: q/k/v f16 (12 MB) + W f16 (0.75 MB)
    const size_t qkvElems = (size_t)Bn * Sn * Dn;
    _Float16* qo = (_Float16*)d_ws;
    _Float16* ko = qo + qkvElems;
    _Float16* vo = ko + qkvElems;
    _Float16* Wh = vo + qkvElems;

    w_to_f16<<<dim3(Dn * En / (256 * 4), 3), 256, 0, stream>>>(Wq, Wk, Wv, Wh);

    qkv_proj<<<dim3(Bn * Sn / 16, 1, 3), 128, 0, stream>>>(
        query, key_in, value, Wh, bq, bk, bv, qo, ko, vo);

    flash_attn<<<dim3(Sn / 64, Bn), 128, 0, stream>>>(qo, ko, vo, out);
}
